// _SConv_58961311039779
// MI455X (gfx1250) — matrix operand order repaired and checked by reading, not compiled, not run on hardware
//
#include <hip/hip_runtime.h>
#include <hip/hip_bf16.h>
#include <stdint.h>

// MI455X / gfx1250: wave32, WMMA f32_16x16x32_bf16, split-precision (bf16x3)
typedef __attribute__((ext_vector_type(16))) __bf16 v16bf;
typedef __attribute__((ext_vector_type(8)))  float  v8f;

#define N_BATCH 32
#define C_IN    128
#define HW      64
#define O_OUT   128
#define NTAPS   5

// LDS pitches in halfwords; row byte pitch must be 16B-aligned for ds_load_b128.
#define ROWA 136   // 272 bytes: bank stride 68 -> conflict-free 16-lane fragment reads
#define ROWB 56    // 112 bytes: 16B aligned, low-conflict

__device__ __forceinline__ unsigned short f32_to_bf16_rne(float f) {
    uint32_t u = __float_as_uint(f);
    uint32_t r = u + 0x7FFFu + ((u >> 16) & 1u);   // round-to-nearest-even
    return (unsigned short)(r >> 16);
}

__device__ __forceinline__ float bf16_bits_to_f32(unsigned short h) {
    return __uint_as_float(((uint32_t)h) << 16);
}

// Split f into hi (bf16) and residual lo (bf16): f ~= hi + lo to ~2^-16 rel.
__device__ __forceinline__ void split_bf16(float f, unsigned short& hi, unsigned short& lo) {
    hi = f32_to_bf16_rne(f);
    float r = f - bf16_bits_to_f32(hi);
    lo = f32_to_bf16_rne(r);
}

__global__ __launch_bounds__(256)
void sconv_wmma_bf16x3(const float* __restrict__ x,
                       const float* __restrict__ w,
                       float* __restrict__ out) {
    // hi/lo planes for weights and x slab
    __shared__ unsigned short ldsAh[O_OUT * ROWA];
    __shared__ unsigned short ldsAl[O_OUT * ROWA];
    __shared__ unsigned short ldsBh[HW * ROWB];
    __shared__ unsigned short ldsBl[HW * ROWB];

    const int b = blockIdx.x;
    const int h = b & 63;
    const int n = (b >> 6) & 31;
    const int j = b >> 11;                         // tap index 0..4

    // Tap offsets (ky-1, kx-1) for SKEL_TAPS = (0,1),(1,0),(1,1),(1,2),(2,1)
    const int dy_tab[NTAPS] = {-1, 0, 0, 0, 1};
    const int dx_tab[NTAPS] = { 0,-1, 0, 1, 0};
    const int dy = dy_tab[j];
    const int dx = dx_tab[j];

    const int t = threadIdx.x;

    // ---- Stage A: weight[o, c, j] -> hi/lo bf16 planes (once per workgroup).
    // weight element (o,c,j) lives at ((o*128 + c)*5 + j).
    #pragma unroll 4
    for (int i = 0; i < 64; ++i) {
        int idx = t + 256 * i;                     // idx = o*128 + c
        float wv = w[(size_t)idx * NTAPS + j];
        unsigned short hi, lo;
        split_bf16(wv, hi, lo);
        int off = (idx >> 7) * ROWA + (idx & 127);
        ldsAh[off] = hi;
        ldsAl[off] = lo;
    }

    const int wave  = t >> 5;                      // 0..7 -> M tile
    const int lane  = t & 31;
    const int lmod  = lane & 15;
    const int khalf = lane >> 4;

    v8f acc[4];
    #pragma unroll
    for (int nt = 0; nt < 4; ++nt) acc[nt] = (v8f){};

    // ---- x slab addressing for this tap (zero padding at edges).
    const int  yy     = h + dy;
    const bool row_ok = (yy >= 0) && (yy < HW);
    const size_t xbase = (size_t)n * C_IN * HW * HW + (size_t)(row_ok ? yy : 0) * HW;

    const int  sw    = t & 63;                     // staged w coordinate
    const int  sc    = t >> 6;                     // 0..3 c sub-lane
    const int  swx   = sw + dx;
    const bool ld_ok = row_ok && (swx >= 0) && (swx < HW);

    for (int kc = 0; kc < C_IN; kc += 32) {
        __syncthreads();                           // slab reuse guard (also covers A stage)

        // Stage B chunk: hi/lo bf16 of x[n, c, yy, w+dx] (0 at pads).
        #pragma unroll
        for (int i = 0; i < 8; ++i) {
            int cc = sc + 4 * i;                   // 0..31
            float xv = ld_ok ? x[xbase + (size_t)(kc + cc) * (HW * HW) + swx] : 0.0f;
            unsigned short hi, lo;
            split_bf16(xv, hi, lo);
            int off = sw * ROWB + cc;
            ldsBh[off] = hi;
            ldsBl[off] = lo;
        }
        // Prefetch next C chunk into cache (global_prefetch_b8 on gfx1250).
        if (ld_ok && (kc + 32 < C_IN))
            __builtin_prefetch(&x[xbase + (size_t)(kc + 32 + sc) * (HW * HW) + swx], 0, 0);

        __syncthreads();

        // A fragments (16x32 bf16, ISA layout: lanes0-15 K{0-7,16-23}, lanes16-31 K{8-15,24-31})
        int ao = wave * 16 + lmod;
        size_t aoff = (size_t)ao * (ROWA * 2) + (size_t)(kc + khalf * 8) * 2;
        v16bf aH, aL;
        ((uint4*)&aH)[0] = *(const uint4*)((const char*)ldsAh + aoff);
        ((uint4*)&aH)[1] = *(const uint4*)((const char*)ldsAh + aoff + 32);
        ((uint4*)&aL)[0] = *(const uint4*)((const char*)ldsAl + aoff);
        ((uint4*)&aL)[1] = *(const uint4*)((const char*)ldsAl + aoff + 32);

        #pragma unroll
        for (int nt = 0; nt < 4; ++nt) {
            // B fragments (32x16 bf16: lanes hold N=w, K halves per lane group)
            int bw = nt * 16 + lmod;
            size_t boff = (size_t)bw * (ROWB * 2) + (size_t)khalf * 16;
            v16bf bH, bL;
            ((uint4*)&bH)[0] = *(const uint4*)((const char*)ldsBh + boff);
            ((uint4*)&bH)[1] = *(const uint4*)((const char*)ldsBh + boff + 32);
            ((uint4*)&bL)[0] = *(const uint4*)((const char*)ldsBl + boff);
            ((uint4*)&bL)[1] = *(const uint4*)((const char*)ldsBl + boff + 32);

            // Split-precision accumulate: hi*hi + hi*lo + lo*hi (lo*lo dropped).
            acc[nt] = __builtin_amdgcn_wmma_f32_16x16x32_bf16(
                false, aH, false, bH, (short)0, acc[nt], false, false);
            acc[nt] = __builtin_amdgcn_wmma_f32_16x16x32_bf16(
                false, aH, false, bL, (short)0, acc[nt], false, false);
            acc[nt] = __builtin_amdgcn_wmma_f32_16x16x32_bf16(
                false, aL, false, bH, (short)0, acc[nt], false, false);
        }
    }

    // ---- Epilogue: ReLU + store. C/D layout: lane%16 = N(w), VGPR r -> M=r (+8 for lanes 16-31).
    const size_t obase = (((size_t)n * (NTAPS * O_OUT) + (size_t)j * O_OUT) * HW + h) * HW;
    #pragma unroll
    for (int nt = 0; nt < 4; ++nt) {
        #pragma unroll
        for (int r = 0; r < 8; ++r) {
            int o = wave * 16 + khalf * 8 + r;
            float v = acc[nt][r];
            out[obase + (size_t)o * (HW * HW) + nt * 16 + lmod] = v > 0.0f ? v : 0.0f;
        }
    }
}

extern "C" void kernel_launch(void* const* d_in, const int* in_sizes, int n_in,
                              void* d_out, int out_size, void* d_ws, size_t ws_size,
                              hipStream_t stream) {
    (void)in_sizes; (void)n_in; (void)out_size; (void)d_ws; (void)ws_size;
    const float* x  = (const float*)d_in[0];   // [32,128,64,64]
    const float* wt = (const float*)d_in[1];   // [128,128,5]
    float* out = (float*)d_out;                // [32,640,64,64]

    dim3 grid(NTAPS * N_BATCH * HW);           // 10240 workgroups: (tap, n, h)
    dim3 block(256);                           // 8 waves of 32
    sconv_wmma_bf16x3<<<grid, block, 0, stream>>>(x, wt, out);
}
